// Attention_52158082843477
// MI455X (gfx1250) — compile-verified, matrix operand order checked by reading, not run on hardware
//
#include <hip/hip_runtime.h>
#include <hip/hip_bf16.h>
#include <cmath>

// ---------------- problem constants ----------------
#define EMB  768
#define TSEQ 4096
#define NB   4
#define HD   64
#define WELE (64 * EMB)          // elements per W matrix
#define LOG2E 1.44269504088896340736f

typedef __attribute__((ext_vector_type(16))) _Float16 v16h;
typedef __attribute__((ext_vector_type(8)))  _Float16 v8h;
typedef __attribute__((ext_vector_type(8)))  float    v8f;

__device__ __forceinline__ v16h pack16(v8h lo, v8h hi) {
    v16h r;
#pragma unroll
    for (int i = 0; i < 8; ++i) { r[i] = lo[i]; r[8 + i] = hi[i]; }
    return r;
}

// reduce across the 16 lanes of each half-wave (lanes 0-15 / 16-31 stay separate)
__device__ __forceinline__ float rmax16(float v) {
#pragma unroll
    for (int m = 1; m < 16; m <<= 1) v = fmaxf(v, __shfl_xor(v, m, 32));
    return v;
}
__device__ __forceinline__ float rsum16(float v) {
#pragma unroll
    for (int m = 1; m < 16; m <<= 1) v += __shfl_xor(v, m, 32);
    return v;
}

// ============================================================
// Stage 0: one-shot W -> f16.  Wq absorbs softmax scale AND
// log2(e) so stage-2 softmax runs natively in the exp2 domain.
// ============================================================
__global__ __launch_bounds__(256)
void convert_w_kernel(const float* __restrict__ Wq,
                      const float* __restrict__ Wk,
                      const float* __restrict__ Wv,
                      _Float16* __restrict__ w16,
                      float qscale)
{
    const int idx = blockIdx.x * 256 + threadIdx.x;      // 0 .. 3*WELE-1
    const int sel = idx / WELE;
    const int rem = idx - sel * WELE;
    float v = (sel == 0) ? Wq[rem] * qscale : (sel == 1) ? Wk[rem] : Wv[rem];
    w16[idx] = (_Float16)v;
}

// ============================================================
// Stage 1: fused q/k/v projection  [B*T,768] x [768,64] -> f16
//   one wave per 16-row tile computes ALL THREE outputs so the
//   f32->f16 x-tile conversion is done once per K-step.
//   v is stored transposed: vT[b, h, t].
// ============================================================
__global__ __launch_bounds__(32)
void proj_qkv_kernel(const float* __restrict__ x,
                     const _Float16* __restrict__ w16,   // [3][64][768] f16, Wq pre-scaled
                     _Float16* __restrict__ q16,
                     _Float16* __restrict__ k16,
                     _Float16* __restrict__ vT16)
{
    const int lane = threadIdx.x;
    const int half = lane >> 4;
    const int ln   = lane & 15;
    const int t0   = blockIdx.x * 16;                    // row tile over flattened B*T

    v8f acc[3][4] = {};                                  // [sel][head-group]

    for (int kc = 0; kc < EMB; kc += 32) {
        const int c0 = kc + half * 8;                    // K halves {0..7,16..23}/{8..15,24..31}
        const float* xp = x + (size_t)(t0 + ln) * EMB + c0;
        v16h a;
#pragma unroll
        for (int i = 0; i < 8; ++i) a[i]     = (_Float16)xp[i];
#pragma unroll
        for (int i = 0; i < 8; ++i) a[8 + i] = (_Float16)xp[16 + i];

#pragma unroll
        for (int sel = 0; sel < 3; ++sel)
#pragma unroll
            for (int hg = 0; hg < 4; ++hg) {
                const _Float16* wp = w16 + (size_t)sel * WELE + (hg * 16 + ln) * EMB + c0;
                const v16h bf = pack16(*(const v8h*)wp, *(const v8h*)(wp + 16));
                acc[sel][hg] = __builtin_amdgcn_wmma_f32_16x16x32_f16(
                    false, a, false, bf, (short)0, acc[sel][hg], false, false);
            }
    }

    // C layout: VGPR r at lanes<16 -> row r, lanes>=16 -> row 8+r; col = ln
    const int moff = half * 8;
#pragma unroll
    for (int hg = 0; hg < 4; ++hg)
#pragma unroll
        for (int r = 0; r < 8; ++r) {
            const int t = t0 + moff + r;                 // tiles never cross batch boundary
            q16[(size_t)t * HD + hg * 16 + ln] = (_Float16)acc[0][hg][r];
            k16[(size_t)t * HD + hg * 16 + ln] = (_Float16)acc[1][hg][r];
            const int b = t / TSEQ, tt = t % TSEQ;
            vT16[((size_t)b * HD + hg * 16 + ln) * TSEQ + tt] = (_Float16)acc[2][hg][r];
        }
}

// ============================================================
// Stage 2: causal flash attention, 64-key tiles, exp2 domain.
//   grid (T/64, B), 128 threads = 4 waves; wave owns 16 query rows.
//   S = Q.K^T (8 WMMAs), one online-softmax update per 64 keys,
//   P (16x64) -> LDS -> two A-fragments, O += P.V (8 WMMAs).
// ============================================================
__global__ __launch_bounds__(128)
void flash_attn_kernel(const _Float16* __restrict__ q16,
                       const _Float16* __restrict__ k16,
                       const _Float16* __restrict__ vT16,
                       float* __restrict__ out)
{
    __shared__ __align__(16) _Float16 plds[4][16][64];   // per-wave P tile (2 KB each)

    const int lane = threadIdx.x & 31;
    const int wave = threadIdx.x >> 5;
    const int half = lane >> 4;
    const int ln   = lane & 15;
    const int moff = half * 8;
    const int b    = blockIdx.y;
    const int q0   = blockIdx.x * 64 + wave * 16;        // wave-uniform
    const int q_hi = q0 + 15;

    const _Float16* kb = k16  + (size_t)b * TSEQ * HD;
    const _Float16* vb = vT16 + (size_t)b * HD * TSEQ;

    // Q fragments (16x64 = two 16x32 A fragments), resident for whole loop
    v16h qa[2];
    {
        const _Float16* qp = q16 + ((size_t)b * TSEQ + q0 + ln) * HD;
#pragma unroll
        for (int df = 0; df < 2; ++df) {
            const int d0 = df * 32 + half * 8;
            qa[df] = pack16(*(const v8h*)(qp + d0), *(const v8h*)(qp + d0 + 16));
        }
    }

    float m_i[8], l_i[8];
    v8f   o[4] = {};
#pragma unroll
    for (int r = 0; r < 8; ++r) { m_i[r] = -1.0e30f; l_i[r] = 0.0f; }

    const int sb = half * 8;                             // K-half sub-block base

    for (int s0 = 0; s0 <= q_hi; s0 += 64) {             // wave-uniform trip count
        // ---- S = Q . K^T : four 16x16 f32 tiles covering 64 keys ----
        v8f s_t[4];
#pragma unroll
        for (int nf = 0; nf < 4; ++nf) {
            const _Float16* kp = kb + (size_t)(s0 + nf * 16 + ln) * HD;
            const v16h kf0 = pack16(*(const v8h*)(kp + sb),      *(const v8h*)(kp + sb + 16));
            const v16h kf1 = pack16(*(const v8h*)(kp + 32 + sb), *(const v8h*)(kp + 32 + sb + 16));
            v8f c = {};
            c = __builtin_amdgcn_wmma_f32_16x16x32_f16(false, qa[0], false, kf0, (short)0, c, false, false);
            c = __builtin_amdgcn_wmma_f32_16x16x32_f16(false, qa[1], false, kf1, (short)0, c, false, false);
            s_t[nf] = c;
        }
        if (s0 + 64 <= q_hi) {                           // CDNA5 global_prefetch_b8
            __builtin_prefetch(kb + (size_t)(s0 + 64 + lane * 2) * HD, 0, 1);
            __builtin_prefetch(vb + (size_t)(lane * 2) * TSEQ + (s0 + 64), 0, 1);
        }

        // ---- causal mask + one online-softmax update per 64 keys (log2 domain) ----
        const bool need_mask = (s0 + 63 > q0);
#pragma unroll
        for (int r = 0; r < 8; ++r) {
            if (need_mask) {
                const int qrow = q0 + moff + r;
#pragma unroll
                for (int nf = 0; nf < 4; ++nf)
                    if (s0 + nf * 16 + ln > qrow) s_t[nf][r] = -1.0e30f;
            }
            float tm = fmaxf(fmaxf(s_t[0][r], s_t[1][r]), fmaxf(s_t[2][r], s_t[3][r]));
            tm = rmax16(tm);
            const float mn = fmaxf(m_i[r], tm);
            const float alpha = __builtin_amdgcn_exp2f(m_i[r] - mn);   // v_exp_f32, no mul
            m_i[r] = mn;
            float rs = 0.0f;
#pragma unroll
            for (int nf = 0; nf < 4; ++nf) {
                const float e = __builtin_amdgcn_exp2f(s_t[nf][r] - mn);
                s_t[nf][r] = e;                          // exp2 in place -> P
                rs += e;
            }
            l_i[r] = l_i[r] * alpha + rsum16(rs);
#pragma unroll
            for (int hg = 0; hg < 4; ++hg) o[hg][r] *= alpha;
        }

        // ---- C-layout P -> LDS -> A-fragments (wave-private, no block barrier) ----
#pragma unroll
        for (int r = 0; r < 8; ++r)
#pragma unroll
            for (int nf = 0; nf < 4; ++nf)
                plds[wave][moff + r][nf * 16 + ln] = (_Float16)s_t[nf][r];
        asm volatile("s_wait_dscnt 0x0" ::: "memory");   // in-wave DS ordering
        const _Float16* pp = &plds[wave][ln][0];
        v16h pa[2];
#pragma unroll
        for (int df = 0; df < 2; ++df) {
            const int d0 = df * 32 + sb;
            pa[df] = pack16(*(const v8h*)(pp + d0), *(const v8h*)(pp + d0 + 16));
        }

        // ---- O += P . V  (V pre-transposed: contiguous along key dim) ----
#pragma unroll
        for (int hg = 0; hg < 4; ++hg) {
            const _Float16* vp = vb + (size_t)(hg * 16 + ln) * TSEQ + s0;
            const v16h vf0 = pack16(*(const v8h*)(vp + sb),      *(const v8h*)(vp + sb + 16));
            const v16h vf1 = pack16(*(const v8h*)(vp + 32 + sb), *(const v8h*)(vp + 32 + sb + 16));
            o[hg] = __builtin_amdgcn_wmma_f32_16x16x32_f16(false, pa[0], false, vf0, (short)0, o[hg], false, false);
            o[hg] = __builtin_amdgcn_wmma_f32_16x16x32_f16(false, pa[1], false, vf1, (short)0, o[hg], false, false);
        }
    }

    // ---- epilogue: normalize (rcp + mul) and store fp32 output [B,T,64] ----
#pragma unroll
    for (int r = 0; r < 8; ++r) {
        const float inv = __builtin_amdgcn_rcpf(l_i[r]);
#pragma unroll
        for (int hg = 0; hg < 4; ++hg)
            out[((size_t)b * TSEQ + q0 + moff + r) * HD + hg * 16 + ln] = o[hg][r] * inv;
    }
}

// ============================================================
extern "C" void kernel_launch(void* const* d_in, const int* in_sizes, int n_in,
                              void* d_out, int out_size, void* d_ws, size_t ws_size,
                              hipStream_t stream)
{
    const float* x  = (const float*)d_in[0];
    const float* Wq = (const float*)d_in[1];
    const float* Wk = (const float*)d_in[2];
    const float* Wv = (const float*)d_in[3];
    float* out = (float*)d_out;

    char* ws = (char*)d_ws;
    const size_t npos = (size_t)NB * TSEQ;               // 16384
    const size_t sect = npos * HD * sizeof(_Float16);    // 2 MB
    _Float16* q16  = (_Float16*)(ws);
    _Float16* k16  = (_Float16*)(ws + sect);
    _Float16* vT16 = (_Float16*)(ws + 2 * sect);
    _Float16* w16  = (_Float16*)(ws + 3 * sect);         // 3*64*768*2 = 288 KB

    // softmax scale faithful to reference (C ** -0.05), plus log2(e) so the
    // whole softmax runs in the exp2 domain natively supported by v_exp_f32.
    const float qscale = powf((float)EMB, -0.05f) * LOG2E;

    convert_w_kernel<<<(3 * WELE) / 256, 256, 0, stream>>>(Wq, Wk, Wv, w16, qscale);

    proj_qkv_kernel<<<(unsigned)(npos / 16), 32, 0, stream>>>(x, w16, q16, k16, vT16);

    dim3 g2(TSEQ / 64, NB);
    flash_attn_kernel<<<g2, 128, 0, stream>>>(q16, k16, vT16, out);
}
